// SwinTransformerBlock_2826088480870
// MI455X (gfx1250) — matrix operand order repaired and checked by reading, not compiled, not run on hardware
//
#include <hip/hip_runtime.h>
#include <math.h>

typedef unsigned short u16;
typedef unsigned int   u32;
typedef __attribute__((ext_vector_type(16))) __bf16 bf16x16;
typedef __attribute__((ext_vector_type(8)))  float  f32x8;
typedef __attribute__((ext_vector_type(4)))  u32    u32x4;
typedef __attribute__((ext_vector_type(4)))  int    i32x4;

// ---------------- problem constants ----------------
#define BB     32
#define HH     56
#define WW_    56
#define CC     256
#define HEADS  8
#define WS     7
#define SHIFT  3
#define NTOK   49                 // tokens per window
#define NWIN   64                 // windows per image
#define MTOT   (BB * HH * WW_)    // 100352 rows for all flat GEMMs
#define HID    1024
#define HDIM   32
#define SCALEF 0.17677669529663687f   // 1/sqrt(32)

// -------- CDNA5 async global->LDS path (guarded; falls back to sync) --------
#if defined(__has_builtin)
#  if __has_builtin(__builtin_amdgcn_global_load_async_to_lds_b128)
#    define HAVE_ASYNC_LDS 1
#  endif
#endif
#ifndef HAVE_ASYNC_LDS
#  define HAVE_ASYNC_LDS 0
#endif

#if HAVE_ASYNC_LDS && defined(__has_builtin)
#  if __has_builtin(__builtin_amdgcn_s_wait_asynccnt)
#    define ASYNC_WAIT() __builtin_amdgcn_s_wait_asynccnt(0)
#  else
#    define ASYNC_WAIT() asm volatile("s_wait_asynccnt 0x0" ::: "memory")
#  endif
#else
#  define ASYNC_WAIT() ((void)0)
#endif

// AS1 (global) / AS3 (LDS) int4-vector pointers, per the builtin's signature
typedef __attribute__((address_space(1))) i32x4 as1_i32x4;
typedef __attribute__((address_space(3))) i32x4 as3_i32x4;

// 16B global -> LDS copy, per-lane. Async (ASYNCcnt-tracked) when available.
__device__ __forceinline__ void g2l_b128(const void* g, void* l) {
#if HAVE_ASYNC_LDS
  // generic LDS pointer low 32 bits == LDS byte offset (aperture rule, ISA 10.2)
  __builtin_amdgcn_global_load_async_to_lds_b128(
      (as1_i32x4*)(unsigned long long)g,
      (as3_i32x4*)(unsigned int)(unsigned long long)l, 0, 0);
#else
  *(u32x4*)l = *(const u32x4*)g;
#endif
}

// f32 -> bf16 round-to-nearest-even
__device__ __forceinline__ u16 f2bf(float f) {
  u32 u = __float_as_uint(f);
  u32 r = u + 0x7FFFu + ((u >> 16) & 1u);
  return (u16)(r >> 16);
}

union FragU { bf16x16 v; u32x4 q[2]; };

__device__ __forceinline__ f32x8 wmma_bf16(bf16x16 a, bf16x16 b, f32x8 c) {
  return __builtin_amdgcn_wmma_f32_16x16x32_bf16(false, a, false, b, (short)0, c,
                                                 false, false);
}

// A-fragment (16x32 bf16), row-major source with leading dim ld (u16 elems).
// lane: m = lane&15; K groups: lanes 0-15 hold K {0..7,16..23}, 16-31 hold {8..15,24..31}.
__device__ __forceinline__ bf16x16 load_frag_a(const u16* p, int ld, int lane) {
  int m  = lane & 15;
  int kb = (lane >> 4) << 3;                // 0 or 8
  const u16* r = p + m * ld;
  FragU f;
  f.q[0] = *(const u32x4*)(r + kb);         // elems 0..7  -> K = kb..kb+7
  f.q[1] = *(const u32x4*)(r + 16 + kb);    // elems 8..15 -> K = 16+kb..
  return f.v;
}

// B-fragment (32x16 bf16) from W-style [n][k] row-major (k contiguous).
// lane: n = lane&15; K groups: lanes 0-15 hold K {0..7,16..23}, 16-31 hold {8..15,24..31}.
__device__ __forceinline__ bf16x16 load_frag_b(const u16* p, int ld, int lane) {
  int n  = lane & 15;
  int k0 = (lane >> 4) << 3;                // 0 or 8
  const u16* r = p + n * ld + k0;
  FragU f;
  f.q[0] = *(const u32x4*)(r);
  f.q[1] = *(const u32x4*)(r + 16);
  return f.v;
}

// windowed (shifted) flat row -> original flat row (bijection; also the LN1 gather map)
__device__ __forceinline__ int win_to_orig(int m) {
  int bw = m / NTOK, n = m - bw * NTOK;
  int b  = bw >> 6, nw = bw & 63;
  int wh = nw >> 3, ww = nw & 7;
  int rr = n / WS,  cc = n - rr * WS;
  int hs = wh * WS + rr;
  int ws = ww * WS + cc;
  int h = hs + SHIFT; if (h >= HH)  h -= HH;
  int w = ws + SHIFT; if (w >= WW_) w -= WW_;
  return b * (HH * WW_) + h * WW_ + w;
}

// ---------------- weight f32 -> bf16 ----------------
__global__ void cvt_bf16_kernel(const float* __restrict__ in, u16* __restrict__ out, int n) {
  int i = blockIdx.x * 256 + threadIdx.x;
  if (i < n) out[i] = f2bf(in[i]);
}

// ---------------- LayerNorm (optionally fused shift+window gather) ----------------
template <bool WINDOWED>
__global__ __launch_bounds__(256)
void ln_kernel(const float* __restrict__ x, const float* __restrict__ g,
               const float* __restrict__ be, u16* __restrict__ out) {
  int wave = threadIdx.x >> 5, lane = threadIdx.x & 31;
  int tok  = blockIdx.x * 8 + wave;                 // exact: 12544*8 == MTOT
  int src  = WINDOWED ? win_to_orig(tok) : tok;
  const float* row = x + (size_t)src * CC;
  float v[8], s = 0.f;
#pragma unroll
  for (int j = 0; j < 8; ++j) { v[j] = row[lane + 32 * j]; s += v[j]; }
#pragma unroll
  for (int o = 16; o > 0; o >>= 1) s += __shfl_xor(s, o, 32);
  float mean = s * (1.f / CC);
  float vs = 0.f;
#pragma unroll
  for (int j = 0; j < 8; ++j) { float d = v[j] - mean; vs += d * d; }
#pragma unroll
  for (int o = 16; o > 0; o >>= 1) vs += __shfl_xor(vs, o, 32);
  float inv = rsqrtf(vs * (1.f / CC) + 1e-5f);
  u16* orow = out + (size_t)tok * CC;
#pragma unroll
  for (int j = 0; j < 8; ++j) {
    int c = lane + 32 * j;
    orow[c] = f2bf((v[j] - mean) * inv * g[c] + be[c]);
  }
}

// ---------------- generic bf16 WMMA GEMM: C[M,N] = A[M,K] * W[N,K]^T + bias ----------------
// 128x128 block tile, 8 waves each owning 32x64, K-step 32, double-buffered LDS
// staged with async global->LDS (ASYNCcnt) when available.
enum { EP_STORE_BF16 = 0, EP_GELU_BF16 = 1, EP_RES_F32 = 2, EP_RESWIN_F32 = 3 };

template <int EP>
__global__ __launch_bounds__(256)
void gemm_kernel(const u16* __restrict__ A, const u16* __restrict__ W,
                 const float* __restrict__ bias, const float* __restrict__ resid,
                 u16* __restrict__ outB, float* __restrict__ outF, int K, int N) {
  __shared__ u16 As[2][128][32];
  __shared__ u16 Bs[2][128][32];
  int t = threadIdx.x;
  int wave = t >> 5, lane = t & 31;
  int wm = wave & 3, wn = wave >> 2;                // 4x2 wave grid -> 128x128 tile
  int m0 = blockIdx.x * 128, n0 = blockIdx.y * 128;

  f32x8 acc[2][4];
#pragma unroll
  for (int i = 0; i < 2; ++i)
#pragma unroll
    for (int j = 0; j < 4; ++j)
#pragma unroll
      for (int r = 0; r < 8; ++r) acc[i][j][r] = 0.f;

  // staging: 256 threads, each moves 32B of A and 32B of B per K-tile
  int sr = t >> 1, sh = (t & 1) << 4;               // row 0..127, col-offset 0/16
  const u16* Ag = A + (size_t)(m0 + sr) * K + sh;
  const u16* Bg = W + (size_t)(n0 + sr) * K + sh;

  int T = K >> 5;
  // prologue: stage tile 0
  {
    g2l_b128(Ag,     &As[0][sr][sh]);
    g2l_b128(Ag + 8, &As[0][sr][sh + 8]);
    g2l_b128(Bg,     &Bs[0][sr][sh]);
    g2l_b128(Bg + 8, &Bs[0][sr][sh + 8]);
    ASYNC_WAIT();
    __syncthreads();
  }

  for (int ti = 0; ti < T; ++ti) {
    int cur = ti & 1;
    if (ti + 1 < T) {
      int k1 = (ti + 1) << 5;
      g2l_b128(Ag + k1,     &As[cur ^ 1][sr][sh]);
      g2l_b128(Ag + k1 + 8, &As[cur ^ 1][sr][sh + 8]);
      g2l_b128(Bg + k1,     &Bs[cur ^ 1][sr][sh]);
      g2l_b128(Bg + k1 + 8, &Bs[cur ^ 1][sr][sh + 8]);
    }
    bf16x16 fa0 = load_frag_a(&As[cur][wm * 32][0],      32, lane);
    bf16x16 fa1 = load_frag_a(&As[cur][wm * 32 + 16][0], 32, lane);
    bf16x16 fb[4];
#pragma unroll
    for (int ni = 0; ni < 4; ++ni)
      fb[ni] = load_frag_b(&Bs[cur][wn * 64 + ni * 16][0], 32, lane);
#pragma unroll
    for (int ni = 0; ni < 4; ++ni) {
      acc[0][ni] = wmma_bf16(fa0, fb[ni], acc[0][ni]);
      acc[1][ni] = wmma_bf16(fa1, fb[ni], acc[1][ni]);
    }
    ASYNC_WAIT();          // next tile fully landed in LDS
    __syncthreads();       // everyone done reading cur before it is re-staged
  }

#pragma unroll
  for (int mi = 0; mi < 2; ++mi) {
    int mg0 = m0 + wm * 32 + mi * 16 + ((lane >> 4) << 3);
#pragma unroll
    for (int ni = 0; ni < 4; ++ni) {
      int col = n0 + wn * 64 + ni * 16 + (lane & 15);
      float bv = bias[col];
#pragma unroll
      for (int r = 0; r < 8; ++r) {
        int m = mg0 + r;
        float v = acc[mi][ni][r] + bv;
        if (EP == EP_STORE_BF16) {
          outB[(size_t)m * N + col] = f2bf(v);
        } else if (EP == EP_GELU_BF16) {
          v = 0.5f * v * (1.f + erff(v * 0.70710678118654752f));
          outB[(size_t)m * N + col] = f2bf(v);
        } else if (EP == EP_RES_F32) {
          outF[(size_t)m * N + col] = v + resid[(size_t)m * N + col];
        } else {                                    // window-reverse + unshift
          int om = win_to_orig(m);
          outF[(size_t)om * CC + col] = v + resid[(size_t)om * CC + col];
        }
      }
    }
  }
}

// ---------------- windowed attention: one wave per (window, head) ----------------
__global__ __launch_bounds__(64)
void attn_kernel(const u16* __restrict__ qkv, const float* __restrict__ relb,
                 const float* __restrict__ mask, u16* __restrict__ ao) {
  __shared__ float Sb[2][64][64];   // scores (f32)
  __shared__ u16   Pb[2][64][64];   // probabilities (bf16)
  __shared__ u16   Vt[2][32][64];   // V transposed: [d][token]
  int wave = threadIdx.x >> 5, lane = threadIdx.x & 31;
  int bw   = blockIdx.x;
  int head = blockIdx.y * 2 + wave;
  int nw   = bw & (NWIN - 1);
  const u16* base = qkv + (size_t)bw * NTOK * 768;
  int qc = head * HDIM, kc = CC + head * HDIM, vc = 2 * CC + head * HDIM;

  // stage V^T (zero-pad tokens 49..63)
  for (int t0 = lane; t0 < 64; t0 += 32) {
    if (t0 < NTOK) {
      const u16* vr = base + (size_t)t0 * 768 + vc;
#pragma unroll
      for (int d = 0; d < HDIM; ++d) Vt[wave][d][t0] = vr[d];
    } else {
#pragma unroll
      for (int d = 0; d < HDIM; ++d) Vt[wave][d][t0] = 0;
    }
  }

  // Q fragments for 4 row blocks (K = head_dim = 32, exactly one WMMA step)
  bf16x16 aq[4];
#pragma unroll
  for (int mb = 0; mb < 4; ++mb) {
    int tok = mb * 16 + (lane & 15);
    int kb  = (lane >> 4) << 3;
    FragU f;
    if (tok < NTOK) {
      const u16* r = base + (size_t)tok * 768 + qc;
      f.q[0] = *(const u32x4*)(r + kb);
      f.q[1] = *(const u32x4*)(r + 16 + kb);
    } else { f.q[0] = 0; f.q[1] = 0; }
    aq[mb] = f.v;
  }

  // S = Q K^T  (B-fragment = rows of K, contiguous head_dim)
  for (int nb = 0; nb < 4; ++nb) {
    int key = nb * 16 + (lane & 15);
    int ke  = (lane >> 4) << 3;
    FragU fb;
    if (key < NTOK) {
      const u16* r = base + (size_t)key * 768 + kc + ke;
      fb.q[0] = *(const u32x4*)(r);
      fb.q[1] = *(const u32x4*)(r + 16);
    } else { fb.q[0] = 0; fb.q[1] = 0; }
#pragma unroll
    for (int mb = 0; mb < 4; ++mb) {
      f32x8 acc;
#pragma unroll
      for (int r = 0; r < 8; ++r) acc[r] = 0.f;
      acc = wmma_bf16(aq[mb], fb.v, acc);
      int col = nb * 16 + (lane & 15);
      int rb  = mb * 16 + ((lane >> 4) << 3);
#pragma unroll
      for (int r = 0; r < 8; ++r) Sb[wave][rb + r][col] = acc[r];
    }
  }
  __syncthreads();

  // softmax(scale*S + rel_pos_bias + shift_mask) row-per-lane (2 rows/lane)
  for (int n0 = lane; n0 < 64; n0 += 32) {
    if (n0 >= NTOK) {
#pragma unroll
      for (int m = 0; m < 64; ++m) Pb[wave][n0][m] = 0;
      continue;
    }
    int ni = n0 / WS, nj = n0 - ni * WS;
    const float* mrow = mask + ((size_t)nw * NTOK + n0) * NTOK;
    float mx = -1e30f;
    int m = 0;
    for (int mi = 0; mi < WS; ++mi)
      for (int mj = 0; mj < WS; ++mj, ++m) {
        int idx = (ni - mi + WS - 1) * (2 * WS - 1) + (nj - mj + WS - 1);
        float s = Sb[wave][n0][m] * SCALEF + relb[idx * HEADS + head] + mrow[m];
        Sb[wave][n0][m] = s;
        mx = fmaxf(mx, s);
      }
    float sum = 0.f;
    for (m = 0; m < NTOK; ++m) { float e = __expf(Sb[wave][n0][m] - mx); Sb[wave][n0][m] = e; sum += e; }
    float inv = 1.f / sum;
    for (m = 0; m < NTOK; ++m) Pb[wave][n0][m] = f2bf(Sb[wave][n0][m] * inv);
    for (m = NTOK; m < 64; ++m) Pb[wave][n0][m] = 0;
  }
  __syncthreads();

  // O = P V  (K = 64 padded tokens, 2 WMMA K-steps; Vt gives contiguous B-frags)
  f32x8 o[4][2];
#pragma unroll
  for (int mb = 0; mb < 4; ++mb)
#pragma unroll
    for (int nb = 0; nb < 2; ++nb)
#pragma unroll
      for (int r = 0; r < 8; ++r) o[mb][nb][r] = 0.f;

#pragma unroll
  for (int ks = 0; ks < 64; ks += 32) {
    bf16x16 ap[4];
#pragma unroll
    for (int mb = 0; mb < 4; ++mb) {
      int row = mb * 16 + (lane & 15);
      int kb  = ks + ((lane >> 4) << 3);
      FragU f;
      f.q[0] = *(const u32x4*)(&Pb[wave][row][kb]);
      f.q[1] = *(const u32x4*)(&Pb[wave][row][kb + 16]);
      ap[mb] = f.v;
    }
#pragma unroll
    for (int nb = 0; nb < 2; ++nb) {
      int d  = nb * 16 + (lane & 15);
      int kb = ks + ((lane >> 4) << 3);
      FragU f;
      f.q[0] = *(const u32x4*)(&Vt[wave][d][kb]);
      f.q[1] = *(const u32x4*)(&Vt[wave][d][kb + 16]);
      bf16x16 bv = f.v;
#pragma unroll
      for (int mb = 0; mb < 4; ++mb) o[mb][nb] = wmma_bf16(ap[mb], bv, o[mb][nb]);
    }
  }

#pragma unroll
  for (int mb = 0; mb < 4; ++mb) {
    int rb = mb * 16 + ((lane >> 4) << 3);
#pragma unroll
    for (int nb = 0; nb < 2; ++nb) {
      int col = head * HDIM + nb * 16 + (lane & 15);
#pragma unroll
      for (int r = 0; r < 8; ++r) {
        int tok = rb + r;
        if (tok < NTOK)
          ao[((size_t)bw * NTOK + tok) * CC + col] = f2bf(o[mb][nb][r]);
      }
    }
  }
}

// ---------------- host-side orchestration ----------------
extern "C" void kernel_launch(void* const* d_in, const int* in_sizes, int n_in,
                              void* d_out, int out_size, void* d_ws, size_t ws_size,
                              hipStream_t stream) {
  const float* x     = (const float*)d_in[0];
  const float* mask  = (const float*)d_in[1];
  const float* ln1w  = (const float*)d_in[2];
  const float* ln1b  = (const float*)d_in[3];
  const float* qkvw  = (const float*)d_in[4];
  const float* qkvb  = (const float*)d_in[5];
  const float* relb  = (const float*)d_in[6];
  const float* projw = (const float*)d_in[7];
  const float* projb = (const float*)d_in[8];
  const float* ln2w  = (const float*)d_in[9];
  const float* ln2b  = (const float*)d_in[10];
  const float* fc1w  = (const float*)d_in[11];
  const float* fc1b  = (const float*)d_in[12];
  const float* fc2w  = (const float*)d_in[13];
  const float* fc2b  = (const float*)d_in[14];
  float* out = (float*)d_out;

  char* ws = (char*)d_ws;
  auto alloc = [&](size_t bytes) -> void* {
    void* p = (void*)ws;
    ws += (bytes + 255) & ~(size_t)255;
    return p;
  };
  u16*   qkvw_b  = (u16*)alloc((size_t)768 * 256 * 2);
  u16*   projw_b = (u16*)alloc((size_t)256 * 256 * 2);
  u16*   fc1w_b  = (u16*)alloc((size_t)1024 * 256 * 2);
  u16*   fc2w_b  = (u16*)alloc((size_t)256 * 1024 * 2);
  u16*   xw      = (u16*)alloc((size_t)MTOT * 256 * 2);    // LN1, windowed order
  u16*   qkvbuf  = (u16*)alloc((size_t)MTOT * 768 * 2);
  u16*   ao      = (u16*)alloc((size_t)MTOT * 256 * 2);    // attn out, windowed order
  float* x1      = (float*)alloc((size_t)MTOT * 256 * 4);  // residual 1, original order
  u16*   h2      = (u16*)alloc((size_t)MTOT * 256 * 2);
  u16*   h3      = (u16*)alloc((size_t)MTOT * 1024 * 2);

  cvt_bf16_kernel<<<(768 * 256 + 255) / 256, 256, 0, stream>>>(qkvw, qkvw_b, 768 * 256);
  cvt_bf16_kernel<<<(256 * 256 + 255) / 256, 256, 0, stream>>>(projw, projw_b, 256 * 256);
  cvt_bf16_kernel<<<(1024 * 256 + 255) / 256, 256, 0, stream>>>(fc1w, fc1w_b, 1024 * 256);
  cvt_bf16_kernel<<<(256 * 1024 + 255) / 256, 256, 0, stream>>>(fc2w, fc2w_b, 256 * 1024);

  // LN1 + cyclic shift + window partition (gather)
  ln_kernel<true><<<MTOT / 8, 256, 0, stream>>>(x, ln1w, ln1b, xw);

  // QKV GEMM: [100352,256] x [768,256]^T
  gemm_kernel<EP_STORE_BF16><<<dim3(MTOT / 128, 768 / 128), 256, 0, stream>>>(
      xw, qkvw_b, qkvb, nullptr, qkvbuf, nullptr, 256, 768);

  // windowed attention (2048 windows, 8 heads; 2 heads per 64-thread block)
  attn_kernel<<<dim3(BB * NWIN, HEADS / 2), 64, 0, stream>>>(qkvbuf, relb, mask, ao);

  // proj GEMM + window-reverse + unshift + residual -> x1 (f32, original order)
  gemm_kernel<EP_RESWIN_F32><<<dim3(MTOT / 128, 256 / 128), 256, 0, stream>>>(
      ao, projw_b, projb, x, nullptr, x1, 256, 256);

  // LN2 (straight rows)
  ln_kernel<false><<<MTOT / 8, 256, 0, stream>>>(x1, ln2w, ln2b, h2);

  // fc1 + exact GELU
  gemm_kernel<EP_GELU_BF16><<<dim3(MTOT / 128, 1024 / 128), 256, 0, stream>>>(
      h2, fc1w_b, fc1b, nullptr, h3, nullptr, 256, 1024);

  // fc2 + residual -> d_out
  gemm_kernel<EP_RES_F32><<<dim3(MTOT / 128, 256 / 128), 256, 0, stream>>>(
      h3, fc2w_b, fc2b, x1, nullptr, out, 1024, 256);
}
